// UnmaskedAttention_31636729102761
// MI455X (gfx1250) — matrix operand order repaired and checked by reading, not compiled, not run on hardware
//
#include <hip/hip_runtime.h>
#include <math.h>

// ---------------------------------------------------------------------------
// MI455X (gfx1250) fused multi-head cross-attention, f16 WMMA pipeline.
//   B=2, H=16, LE=LD=2048, DE=DD=1024, KQ=V=64
// Workspace layout:
//   Qws  [B,H,LD,64] f16 (pre-scaled by 0.125*log2e)  8 MB @ 0
//   Kws  [B,H,LE,64] f16                              8 MB @ 8 MB
//   Vtws [B,H,64,LE] f16 (transposed)                 8 MB @ 16 MB
//   Ows  [B*LD, H*64] f16                             8 MB @ 24 MB
//   bsum [1024] f32                                   4 KB @ 32 MB
// ---------------------------------------------------------------------------

typedef __attribute__((ext_vector_type(16))) _Float16 v16h;
typedef __attribute__((ext_vector_type(8)))  _Float16 v8h;
typedef __attribute__((ext_vector_type(4)))  _Float16 v4h;
typedef __attribute__((ext_vector_type(8)))  float    v8f;
typedef __attribute__((ext_vector_type(4)))  float    v4f;

#define LOG2E 1.44269504088896340736f

static __device__ inline v16h cat8(v8h lo, v8h hi) {
  return __builtin_shufflevector(lo, hi, 0,1,2,3,4,5,6,7,8,9,10,11,12,13,14,15);
}

// A-fragment (16x32 f16, ISA 7.12.2): lane holds row m=lane&15,
// halves [kb..kb+7] and [kb+16..kb+23], kb = (lane>=16 ? 8 : 0).
static __device__ inline v16h lda_frag(const _Float16* rowbase, int stride) {
  const int lane = threadIdx.x & 31;
  const _Float16* p = rowbase + (lane & 15) * stride + ((lane >> 4) << 3);
  return cat8(*(const v8h*)p, *(const v8h*)(p + 16));
}

// B-fragment (32x16 f16) from an n-major buffer (one row per output column):
// lane holds column n=lane&15, halves [kb..kb+7] and [kb+16..kb+23], kb=(lane>=16?8:0).
static __device__ inline v16h ldb_frag(const _Float16* nbase, int stride) {
  const int lane = threadIdx.x & 31;
  const _Float16* p = nbase + (lane & 15) * stride + ((lane >> 4) << 3);
  return cat8(*(const v8h*)p, *(const v8h*)(p + 16));
}

static __device__ inline v8f wmma_f16(v16h a, v16h b, v8f c) {
  return __builtin_amdgcn_wmma_f32_16x16x32_f16(false, a, false, b,
                                                (short)0, c, false, false);
}

// ---------------------------------------------------------------------------
// bsum[n] = sum_h bo[h,n]
// ---------------------------------------------------------------------------
__global__ void bias_sum_kernel(const float* __restrict__ bo,
                                float* __restrict__ bsum) {
  int i = blockIdx.x * blockDim.x + threadIdx.x;
  if (i < 1024) {
    float s = 0.f;
#pragma unroll
    for (int h = 0; h < 16; ++h) s += bo[h * 1024 + i];
    bsum[i] = s;
  }
}

// ---------------------------------------------------------------------------
// 4096x1024x1024 GEMM, f16 WMMA, f32 accumulate, + per-column bias.
// Templated on LAYOUT / AF16 so the hot loop has no runtime branches.
//   LAYOUT 0: O16 [B,H,L,64];  1: O16 [B,H,64,L];  2: O32 [4096,1024]
//   W: LAYOUT!=2 -> per-head [16,1024,64] heads along N; LAYOUT==2 -> flat
// 256 threads = 8 waves; 128x64 tile; K-step 32; double-buffered LDS,
// software-pipelined global->LDS staging (vector loads/stores).
// wscale is folded into W and bias during f32->f16 conversion.
// ---------------------------------------------------------------------------
template <int LAYOUT, bool AF16>
__global__ __launch_bounds__(256) void gemm_proj_kernel(
    const float* __restrict__ A32, const _Float16* __restrict__ A16,
    const float* __restrict__ W, const float* __restrict__ bias, float wscale,
    _Float16* __restrict__ O16, float* __restrict__ O32) {
  __shared__ _Float16 As[2][128 * 40];  // [m][k], row stride 40 halves (80B)
  __shared__ _Float16 Bs[2][64 * 40];   // transposed: [n][k]

  const int t = threadIdx.x;
  const int wave = t >> 5, lane = t & 31, lo = lane & 15, hi = lane >> 4;
  const int m0 = blockIdx.x * 128;
  const int n0 = blockIdx.y * 64;

  // staging geometry (compile-time per thread)
  // A f32: 4 x v4f   idx=i*256+t : m=idx>>3,  k=(idx&7)*4
  // A f16: 2 x v8h   idx=i*256+t : m=idx>>2,  k=(idx&3)*8
  // B f32: 2 x v4f   idx=i*256+t : kl=idx>>4, nl=(idx&15)*4
  v4f ra[4];
  v8h ra16[2];
  v4f rb[2];

  auto load_a = [&](int kk) {
    if (AF16) {
#pragma unroll
      for (int i = 0; i < 2; ++i) {
        int idx = i * 256 + t, m = idx >> 2, k = (idx & 3) << 3;
        ra16[i] = *(const v8h*)(A16 + (size_t)(m0 + m) * 1024 + kk + k);
      }
    } else {
#pragma unroll
      for (int i = 0; i < 4; ++i) {
        int idx = i * 256 + t, m = idx >> 3, k = (idx & 7) << 2;
        ra[i] = *(const v4f*)(A32 + (size_t)(m0 + m) * 1024 + kk + k);
      }
    }
  };
  auto load_b = [&](int kk) {
#pragma unroll
    for (int i = 0; i < 2; ++i) {
      int idx = i * 256 + t, kl = idx >> 4, nl = (idx & 15) << 2;
      int n = n0 + nl, kg = kk + kl;
      const float* wp = (LAYOUT == 2)
                            ? W + (size_t)kg * 1024 + n
                            : W + ((size_t)(n >> 6) * 1024 + kg) * 64 + (n & 63);
      rb[i] = *(const v4f*)wp;
    }
  };
  auto store_stage = [&](int buf) {
    if (AF16) {
#pragma unroll
      for (int i = 0; i < 2; ++i) {
        int idx = i * 256 + t, m = idx >> 2, k = (idx & 3) << 3;
        *(v8h*)&As[buf][m * 40 + k] = ra16[i];
      }
    } else {
#pragma unroll
      for (int i = 0; i < 4; ++i) {
        int idx = i * 256 + t, m = idx >> 3, k = (idx & 7) << 2;
        v4h h4;
#pragma unroll
        for (int j = 0; j < 4; ++j) h4[j] = (_Float16)ra[i][j];
        *(v4h*)&As[buf][m * 40 + k] = h4;
      }
    }
#pragma unroll
    for (int i = 0; i < 2; ++i) {
      int idx = i * 256 + t, kl = idx >> 4, nl = (idx & 15) << 2;
#pragma unroll
      for (int j = 0; j < 4; ++j)
        Bs[buf][(nl + j) * 40 + kl] = (_Float16)(rb[i][j] * wscale);
    }
  };

  v8f acc[4] = {};

  load_a(0);
  load_b(0);
  store_stage(0);

  for (int kt = 0; kt < 32; ++kt) {
    __syncthreads();
    const int cur = kt & 1;
    if (kt + 1 < 32) {  // prefetch next tile while computing this one
      load_a((kt + 1) * 32);
      load_b((kt + 1) * 32);
    }
    v16h af = lda_frag(&As[cur][wave * 16 * 40], 40);
#pragma unroll
    for (int j = 0; j < 4; ++j) {
      v16h bf = ldb_frag(&Bs[cur][j * 16 * 40], 40);
      acc[j] = wmma_f16(af, bf, acc[j]);
    }
    if (kt + 1 < 32) store_stage(cur ^ 1);
  }

  // epilogue: C/D layout -> lane hi-half owns rows r+8*hi, column lane&15
#pragma unroll
  for (int j = 0; j < 4; ++j) {
    int n = n0 + j * 16 + lo;
    float bv = bias[n] * wscale;
#pragma unroll
    for (int r = 0; r < 8; ++r) {
      int m = m0 + wave * 16 + r + 8 * hi;
      float val = acc[j][r] + bv;
      if (LAYOUT == 2) {
        O32[(size_t)m * 1024 + n] = val;
      } else {
        int b = m >> 11, l = m & 2047;
        int h = n >> 6, c = n & 63;
        size_t base = (size_t)(b * 16 + h);
        if (LAYOUT == 0)
          O16[(base * 2048 + l) * 64 + c] = (_Float16)val;  // [B,H,L,64]
        else
          O16[(base * 64 + c) * 2048 + l] = (_Float16)val;  // [B,H,64,L]
      }
    }
  }
}

// ---------------------------------------------------------------------------
// Flash attention: one WG per (b, h, 128-row Q block); 8 waves x 16 Q rows.
// Q was pre-scaled by 0.125*log2e, so softmax runs in base-2 directly.
// ---------------------------------------------------------------------------
__global__ __launch_bounds__(256) void attention_kernel(
    const _Float16* __restrict__ Q, const _Float16* __restrict__ K,
    const _Float16* __restrict__ Vt, _Float16* __restrict__ O) {
  __shared__ _Float16 Pl[8 * 16 * 72];  // per-wave 16x64 P strip, stride 72

  const int t = threadIdx.x;
  const int wave = t >> 5, lane = t & 31, lo = lane & 15, hi = lane >> 4;
  const int blk = blockIdx.x;
  const int qb = blk & 15;        // 16 Q blocks of 128 rows
  const int h = (blk >> 4) & 15;  // 16 heads
  const int b = blk >> 8;         // 2 batches
  const int bh = b * 16 + h;

  const _Float16* Qbh = Q + (size_t)bh * 2048 * 64;
  const _Float16* Kbh = K + (size_t)bh * 2048 * 64;
  const _Float16* Vbh = Vt + (size_t)bh * 64 * 2048;
  const int qrow0 = qb * 128 + wave * 16;

  v16h qf0 = lda_frag(Qbh + (size_t)qrow0 * 64 + 0, 64);
  v16h qf1 = lda_frag(Qbh + (size_t)qrow0 * 64 + 32, 64);

  v8f o[4] = {};
  float rmax[8], rsum[8];
#pragma unroll
  for (int r = 0; r < 8; ++r) { rmax[r] = -3.0e38f; rsum[r] = 0.f; }

  _Float16* pw = Pl + wave * 16 * 72;

  for (int kt = 0; kt < 32; ++kt) {  // 32 key tiles of 64
    v8f s[4] = {};
#pragma unroll
    for (int ks = 0; ks < 2; ++ks) {
      const v16h qf = ks ? qf1 : qf0;
#pragma unroll
      for (int j = 0; j < 4; ++j) {
        v16h bf = ldb_frag(Kbh + (size_t)(kt * 64 + j * 16) * 64 + ks * 32, 64);
        s[j] = wmma_f16(qf, bf, s[j]);
      }
    }
    // online softmax (base-2); row r lives across a 16-lane half
#pragma unroll
    for (int r = 0; r < 8; ++r) {
      float tm = fmaxf(fmaxf(s[0][r], s[1][r]), fmaxf(s[2][r], s[3][r]));
#pragma unroll
      for (int off = 1; off < 16; off <<= 1)
        tm = fmaxf(tm, __shfl_xor(tm, off, 32));
      float mnew = fmaxf(rmax[r], tm);
      float alpha = exp2f(rmax[r] - mnew);
      rmax[r] = mnew;
      float psum = 0.f;
#pragma unroll
      for (int j = 0; j < 4; ++j) {
        float p = exp2f(s[j][r] - mnew);
        psum += p;
        pw[(r + 8 * hi) * 72 + j * 16 + lo] = (_Float16)p;
        o[j][r] *= alpha;
      }
#pragma unroll
      for (int off = 1; off < 16; off <<= 1)
        psum += __shfl_xor(psum, off, 32);
      rsum[r] = rsum[r] * alpha + psum;
    }
    // O += P @ V  (P re-read from LDS in A-layout; Vt rows are n-major)
    v16h pf0 = lda_frag(pw, 72);
    v16h pf1 = lda_frag(pw + 32, 72);
#pragma unroll
    for (int ks = 0; ks < 2; ++ks) {
      const v16h pf = ks ? pf1 : pf0;
#pragma unroll
      for (int j = 0; j < 4; ++j) {
        v16h vf = ldb_frag(Vbh + (size_t)(j * 16) * 2048 + kt * 64 + ks * 32,
                           2048);
        o[j] = wmma_f16(pf, vf, o[j]);
      }
    }
  }

  // normalize and store O as [B*L, H*64] f16 (A matrix for output GEMM)
#pragma unroll
  for (int j = 0; j < 4; ++j)
#pragma unroll
    for (int r = 0; r < 8; ++r) {
      float val = o[j][r] / rsum[r];
      int q = qrow0 + r + 8 * hi;
      int col = h * 64 + j * 16 + lo;
      O[((size_t)(b * 2048 + q)) * 1024 + col] = (_Float16)val;
    }
}

// ---------------------------------------------------------------------------
extern "C" void kernel_launch(void* const* d_in, const int* in_sizes, int n_in,
                              void* d_out, int out_size, void* d_ws,
                              size_t ws_size, hipStream_t stream) {
  (void)in_sizes; (void)n_in; (void)out_size; (void)ws_size;
  const float* enc = (const float*)d_in[0];
  const float* dec = (const float*)d_in[1];
  const float* Wq  = (const float*)d_in[2];
  const float* bq  = (const float*)d_in[3];
  const float* Wk  = (const float*)d_in[4];
  const float* bk  = (const float*)d_in[5];
  const float* Wv  = (const float*)d_in[6];
  const float* bv  = (const float*)d_in[7];
  const float* Wo  = (const float*)d_in[8];
  const float* bo  = (const float*)d_in[9];
  float* out = (float*)d_out;

  char* ws = (char*)d_ws;
  const size_t MB = 1u << 20;
  _Float16* Qws  = (_Float16*)(ws + 0 * MB);
  _Float16* Kws  = (_Float16*)(ws + 8 * MB);
  _Float16* Vtws = (_Float16*)(ws + 16 * MB);
  _Float16* Ows  = (_Float16*)(ws + 24 * MB);
  float*    bsum = (float*)(ws + 32 * MB);

  bias_sum_kernel<<<dim3(4), dim3(256), 0, stream>>>(bo, bsum);

  dim3 gg(32, 16), gb(256);
  const float qs = 0.125f * LOG2E;  // 1/sqrt(64) * log2e folded into Q
  gemm_proj_kernel<0, false><<<gg, gb, 0, stream>>>(dec, nullptr, Wq, bq, qs,
                                                    Qws, nullptr);
  gemm_proj_kernel<0, false><<<gg, gb, 0, stream>>>(enc, nullptr, Wk, bk, 1.f,
                                                    Kws, nullptr);
  gemm_proj_kernel<1, false><<<gg, gb, 0, stream>>>(enc, nullptr, Wv, bv, 1.f,
                                                    Vtws, nullptr);

  attention_kernel<<<dim3(512), gb, 0, stream>>>(Qws, Kws, Vtws, Ows);

  gemm_proj_kernel<2, true><<<gg, gb, 0, stream>>>(nullptr, Ows, Wo, bsum, 1.f,
                                                   nullptr, out);
}
